// NemaFlowModel_69492570849883
// MI455X (gfx1250) — compile-verified, matrix operand order checked by reading, not run on hardware
//
#include <hip/hip_runtime.h>
#include <math.h>

// Model dims
#define B_   4
#define L_   1024
#define D_   1024
#define H_   16
#define HD_  64
#define NL_  8
#define CD_  512
#define KV_  64          // discrete vocab K
#define RFF_ 256
#define FF_  2752
#define ADA_ (6*D_)

typedef __attribute__((ext_vector_type(16))) __bf16 v16bf;
typedef __attribute__((ext_vector_type(8)))  float  v8f;

union FragBF { uint4 q[2]; v16bf v; };

__device__ __forceinline__ unsigned short f2bf(float x) {
  union { float f; unsigned int u; } c; c.f = x;
  unsigned int u = c.u;
  unsigned int r = (u + 0x7FFFu + ((u >> 16) & 1u)) >> 16;
  return (unsigned short)r;
}
__device__ __forceinline__ float siluf_(float x) { return x / (1.f + __expf(-x)); }

#define NEG_INF (-__builtin_inff())

// ---------------------------------------------------------------------------
// Weight convert + transpose: W fp32 [K][N] -> Wt bf16 [N][K]
// grid = (K/32, N/32), block = (32,8). K,N multiples of 32 for all weights.
// ---------------------------------------------------------------------------
__global__ __launch_bounds__(256)
void k_wt(const float* __restrict__ W, int N, int K, unsigned short* __restrict__ Wt)
{
  __shared__ float tile[32][33];
  const int tx = threadIdx.x, ty = threadIdx.y;
  const int k0 = blockIdx.x * 32, n0 = blockIdx.y * 32;
  #pragma unroll
  for (int j = 0; j < 4; j++)
    tile[ty + j * 8][tx] = W[(size_t)(k0 + ty + j * 8) * N + n0 + tx];
  __syncthreads();
  #pragma unroll
  for (int j = 0; j < 4; j++)
    Wt[(size_t)(n0 + ty + j * 8) * K + k0 + tx] = f2bf(tile[tx][ty + j * 8]);
}

// ---------------------------------------------------------------------------
// WMMA GEMM, bf16 A [M][lda] x bf16 Wt [N][K] -> fp32 C [M][ldc] (+bias).
// Block tile 128x128x32, 256 threads = 8 waves, wave tile 64x32 (4x2 frags).
// Global->LDS staging via async DMA (global_load_async_to_lds_b128),
// double-buffered: stage i+1 DMA overlaps stage i WMMAs.
// Requires: M % 128 == 0, K % 32 == 0, Wt rows padded to multiple of 128.
// ---------------------------------------------------------------------------
__global__ __launch_bounds__(256)
void k_gemm_bf(const unsigned short* __restrict__ A, int lda,
               const unsigned short* __restrict__ Wt,
               const float* __restrict__ bias,
               float* __restrict__ C, int ldc,
               int M, int N, int K)
{
  __shared__ unsigned short As[2][128 * 32];
  __shared__ unsigned short Bs[2][128 * 32];

  const int tid  = threadIdx.x;
  const int wave = tid >> 5;
  const int lane = tid & 31;
  const int half = lane >> 4;
  const int l16  = lane & 15;
  const int wrow = wave >> 2;   // 0..1
  const int wcol = wave & 3;    // 0..3
  const int m0 = blockIdx.y * 128;
  const int n0 = blockIdx.x * 128;

  // per-thread staging chunk ids: 512 x 16B chunks per tile, 2 per thread
  const int c0row = tid >> 2, c0kc = tid & 3;            // chunk tid
  const int c1row = (tid + 256) >> 2, c1kc = tid & 3;    // chunk tid+256

  v8f acc[4][2];
  #pragma unroll
  for (int i = 0; i < 4; i++)
    #pragma unroll
    for (int j = 0; j < 2; j++)
      #pragma unroll
      for (int e = 0; e < 8; e++) acc[i][j][e] = 0.f;

  const unsigned aL0 = (unsigned)(size_t)&As[0][0];
  const unsigned aL1 = (unsigned)(size_t)&As[1][0];
  const unsigned bL0 = (unsigned)(size_t)&Bs[0][0];
  const unsigned bL1 = (unsigned)(size_t)&Bs[1][0];

#define STAGE(buf, k0v)                                                          \
  do {                                                                           \
    const unsigned aL = (buf) ? aL1 : aL0;                                       \
    const unsigned bL = (buf) ? bL1 : bL0;                                       \
    unsigned aoff0 = ((unsigned)(m0 + c0row) * (unsigned)lda + (unsigned)(k0v)) * 2u + c0kc * 16u; \
    unsigned al0   = aL + c0row * 64u + c0kc * 16u;                              \
    asm volatile("global_load_async_to_lds_b128 %0, %1, %2"                      \
                 :: "v"(al0), "v"(aoff0), "s"(A) : "memory");                    \
    unsigned aoff1 = ((unsigned)(m0 + c1row) * (unsigned)lda + (unsigned)(k0v)) * 2u + c1kc * 16u; \
    unsigned al1   = aL + c1row * 64u + c1kc * 16u;                              \
    asm volatile("global_load_async_to_lds_b128 %0, %1, %2"                      \
                 :: "v"(al1), "v"(aoff1), "s"(A) : "memory");                    \
    unsigned boff0 = ((unsigned)(n0 + c0row) * (unsigned)K + (unsigned)(k0v)) * 2u + c0kc * 16u;   \
    unsigned bl0   = bL + c0row * 64u + c0kc * 16u;                              \
    asm volatile("global_load_async_to_lds_b128 %0, %1, %2"                      \
                 :: "v"(bl0), "v"(boff0), "s"(Wt) : "memory");                   \
    unsigned boff1 = ((unsigned)(n0 + c1row) * (unsigned)K + (unsigned)(k0v)) * 2u + c1kc * 16u;   \
    unsigned bl1   = bL + c1row * 64u + c1kc * 16u;                              \
    asm volatile("global_load_async_to_lds_b128 %0, %1, %2"                      \
                 :: "v"(bl1), "v"(boff1), "s"(Wt) : "memory");                   \
  } while (0)

  STAGE(0, 0);
  const int nst = K >> 5;
  for (int it = 0; it < nst; ++it) {
    const int nk = (it + 1) << 5;
    if (nk < K) {
      STAGE((it + 1) & 1, nk);
      // 4 async ops just issued for next stage; wait until only those remain
      asm volatile("s_wait_asynccnt 0x4" ::: "memory");
    } else {
      asm volatile("s_wait_asynccnt 0x0" ::: "memory");
    }
    __syncthreads();

    const unsigned short* as = &As[it & 1][0];
    const unsigned short* bs = &Bs[it & 1][0];
    FragBF a[4], b[2];
    #pragma unroll
    for (int mi = 0; mi < 4; mi++) {
      const unsigned short* p = &as[(wrow * 64 + mi * 16 + l16) * 32 + half * 8];
      a[mi].q[0] = *(const uint4*)p;
      a[mi].q[1] = *(const uint4*)(p + 16);
    }
    #pragma unroll
    for (int ni = 0; ni < 2; ni++) {
      const unsigned short* p = &bs[(wcol * 32 + ni * 16 + l16) * 32 + half * 8];
      b[ni].q[0] = *(const uint4*)p;
      b[ni].q[1] = *(const uint4*)(p + 16);
    }
    #pragma unroll
    for (int mi = 0; mi < 4; mi++)
      #pragma unroll
      for (int ni = 0; ni < 2; ni++)
        acc[mi][ni] = __builtin_amdgcn_wmma_f32_16x16x32_bf16(
            false, a[mi].v, false, b[ni].v, (short)0, acc[mi][ni], false, false);
    __syncthreads();
  }
#undef STAGE

  // epilogue
  #pragma unroll
  for (int mi = 0; mi < 4; mi++) {
    #pragma unroll
    for (int ni = 0; ni < 2; ni++) {
      const int col = n0 + wcol * 32 + ni * 16 + l16;
      if (col < N) {
        const float bv = bias ? bias[col] : 0.f;
        #pragma unroll
        for (int i = 0; i < 8; i++) {
          const int row = m0 + wrow * 64 + mi * 16 + half * 8 + i;
          if (row < M) C[(size_t)row * ldc + col] = acc[mi][ni][i] + bv;
        }
      }
    }
  }
}

// ---------------------------------------------------------------------------
// Flash attention with WMMA. Q,K bf16 [bh][l][hd] (roped), V bf16 transposed
// [bh][hd][l]. 128 threads = 4 waves, each wave owns one 16-row q tile.
// Output written as bf16 (B,L,H*HD) feeding the attn-out GEMM.
// ---------------------------------------------------------------------------
__global__ __launch_bounds__(128)
void k_attn(const unsigned short* __restrict__ Qb,
            const unsigned short* __restrict__ Kb,
            const unsigned short* __restrict__ Vt,
            const float* __restrict__ lbias,        // (B,L,L)
            const unsigned char* __restrict__ pm,   // (B,L) bool
            unsigned short* __restrict__ O)         // (B,L,H*HD) bf16
{
  __shared__ unsigned short Pst[4][16 * 40];
  const int tid  = threadIdx.x;
  const int wave = tid >> 5;
  const int lane = tid & 31;
  const int half = lane >> 4;
  const int l16  = lane & 15;
  const int bh = blockIdx.y;
  const int b  = bh / H_;
  const int h  = bh % H_;
  const int q0 = blockIdx.x * 64 + wave * 16;
  const size_t headOff = (size_t)bh * L_ * HD_;

  FragBF aq[2];
  {
    const unsigned short* qp = Qb + headOff + (size_t)(q0 + l16) * HD_;
    aq[0].q[0] = *(const uint4*)(qp + half * 8);
    aq[0].q[1] = *(const uint4*)(qp + 16 + half * 8);
    aq[1].q[0] = *(const uint4*)(qp + 32 + half * 8);
    aq[1].q[1] = *(const uint4*)(qp + 48 + half * 8);
  }

  v8f oacc[4];
  #pragma unroll
  for (int n = 0; n < 4; n++)
    #pragma unroll
    for (int i = 0; i < 8; i++) oacc[n][i] = 0.f;
  float mrow[8], lrow[8];
  #pragma unroll
  for (int i = 0; i < 8; i++) { mrow[i] = NEG_INF; lrow[i] = 0.f; }

  unsigned short* P = Pst[wave];

  for (int k0 = 0; k0 < L_; k0 += 32) {
    v8f s[2];
    #pragma unroll
    for (int jj = 0; jj < 2; jj++) {
      const unsigned short* kp = Kb + headOff + (size_t)(k0 + jj * 16 + l16) * HD_;
      FragBF b0, b1;
      b0.q[0] = *(const uint4*)(kp + half * 8);
      b0.q[1] = *(const uint4*)(kp + 16 + half * 8);
      b1.q[0] = *(const uint4*)(kp + 32 + half * 8);
      b1.q[1] = *(const uint4*)(kp + 48 + half * 8);
      v8f z;
      #pragma unroll
      for (int e = 0; e < 8; e++) z[e] = 0.f;
      z = __builtin_amdgcn_wmma_f32_16x16x32_bf16(false, aq[0].v, false, b0.v, (short)0, z, false, false);
      z = __builtin_amdgcn_wmma_f32_16x16x32_bf16(false, aq[1].v, false, b1.v, (short)0, z, false, false);
      s[jj] = z;
    }
    #pragma unroll
    for (int jj = 0; jj < 2; jj++) {
      const int col = k0 + jj * 16 + l16;
      const bool valid = pm[b * L_ + col] != 0;
      const float* bp = lbias + ((size_t)b * L_ + (q0 + half * 8)) * L_ + col;
      #pragma unroll
      for (int i = 0; i < 8; i++) {
        const float v = s[jj][i] * 0.125f + bp[(size_t)i * L_];
        s[jj][i] = valid ? v : NEG_INF;
      }
    }
    float pnew[2][8];
    #pragma unroll
    for (int i = 0; i < 8; i++) {
      float rm = fmaxf(s[0][i], s[1][i]);
      #pragma unroll
      for (int off = 1; off < 16; off <<= 1) rm = fmaxf(rm, __shfl_xor(rm, off, 32));
      const float mnew = fmaxf(mrow[i], rm);
      const float corr = (mnew == NEG_INF) ? 1.f : __expf(mrow[i] - mnew);
      const float p0 = (s[0][i] == NEG_INF) ? 0.f : __expf(s[0][i] - mnew);
      const float p1 = (s[1][i] == NEG_INF) ? 0.f : __expf(s[1][i] - mnew);
      float rs = p0 + p1;
      #pragma unroll
      for (int off = 1; off < 16; off <<= 1) rs += __shfl_xor(rs, off, 32);
      lrow[i] = lrow[i] * corr + rs;
      mrow[i] = mnew;
      #pragma unroll
      for (int n = 0; n < 4; n++) oacc[n][i] *= corr;
      pnew[0][i] = p0; pnew[1][i] = p1;
    }
    #pragma unroll
    for (int jj = 0; jj < 2; jj++)
      #pragma unroll
      for (int i = 0; i < 8; i++)
        P[(i + 8 * half) * 40 + jj * 16 + l16] = f2bf(pnew[jj][i]);
    FragBF ap;
    {
      const unsigned short* pp = &P[l16 * 40 + half * 8];
      ap.q[0] = *(const uint4*)pp;
      ap.q[1] = *(const uint4*)(pp + 16);
    }
    #pragma unroll
    for (int n = 0; n < 4; n++) {
      const unsigned short* vp = Vt + headOff + (size_t)(n * 16 + l16) * L_ + k0;
      FragBF bv;
      bv.q[0] = *(const uint4*)(vp + half * 8);
      bv.q[1] = *(const uint4*)(vp + 16 + half * 8);
      oacc[n] = __builtin_amdgcn_wmma_f32_16x16x32_bf16(
          false, ap.v, false, bv.v, (short)0, oacc[n], false, false);
    }
  }

  #pragma unroll
  for (int n = 0; n < 4; n++) {
    const int hd = n * 16 + l16;
    #pragma unroll
    for (int i = 0; i < 8; i++) {
      const int row = q0 + half * 8 + i;
      O[((size_t)b * L_ + row) * (H_ * HD_) + h * HD_ + hd] = f2bf(oacc[n][i] / lrow[i]);
    }
  }
}

// ---------------------------------------------------------------------------
// RoPE + layout conversion: qkv f32 (B,L,3,H,HD) -> Q,K bf16 [bh][l][hd],
// V bf16 transposed [bh][hd][l]
// ---------------------------------------------------------------------------
__global__ void k_rope(const float* __restrict__ qkv,
                       unsigned short* __restrict__ Qb,
                       unsigned short* __restrict__ Kb,
                       unsigned short* __restrict__ Vt)
{
  const int idx = blockIdx.x * blockDim.x + threadIdx.x;
  if (idx >= B_ * H_ * L_) return;
  const int l  = idx % L_;
  const int bh = idx / L_;
  const int h  = bh % H_;
  const int b  = bh / H_;
  const float* base = qkv + (size_t)(b * L_ + l) * 3 * H_ * HD_;
  const float* q = base + h * HD_;
  const float* k = base + H_ * HD_ + h * HD_;
  const float* v = base + 2 * H_ * HD_ + h * HD_;
  unsigned short* qo = Qb + ((size_t)bh * L_ + l) * HD_;
  unsigned short* ko = Kb + ((size_t)bh * L_ + l) * HD_;
  for (int d = 0; d < 32; d++) {
    const float inv = __powf(10000.f, -(float)d / 32.f);
    float sn, cs;
    __sincosf((float)l * inv, &sn, &cs);
    const float q1 = q[d], q2 = q[d + 32];
    qo[d]      = f2bf(q1 * cs - q2 * sn);
    qo[d + 32] = f2bf(q2 * cs + q1 * sn);
    const float k1 = k[d], k2 = k[d + 32];
    ko[d]      = f2bf(k1 * cs - k2 * sn);
    ko[d + 32] = f2bf(k2 * cs + k1 * sn);
  }
  unsigned short* vo = Vt + (size_t)bh * HD_ * L_;
  for (int d = 0; d < 64; d++) vo[(size_t)d * L_ + l] = f2bf(v[d]);
}

// ---------------------------------------------------------------------------
// LayerNorm + adaLN modulation: out = bf16( LN(x)*(1+scale[b])+shift[b] )
// ---------------------------------------------------------------------------
__global__ __launch_bounds__(256)
void k_ln_mod(const float* __restrict__ x, unsigned short* __restrict__ out,
              const float* __restrict__ ada, int scale_chunk, int shift_chunk)
{
  __shared__ float red[2][8];
  const int row = blockIdx.x;
  const int b   = row / L_;
  const float* xr = x + (size_t)row * D_;
  float vals[4];
  float s = 0.f, s2 = 0.f;
  #pragma unroll
  for (int j = 0; j < 4; j++) {
    const float v = xr[threadIdx.x + j * 256];
    vals[j] = v; s += v; s2 += v * v;
  }
  #pragma unroll
  for (int off = 1; off < 32; off <<= 1) {
    s  += __shfl_xor(s, off, 32);
    s2 += __shfl_xor(s2, off, 32);
  }
  const int wave = threadIdx.x >> 5;
  if ((threadIdx.x & 31) == 0) { red[0][wave] = s; red[1][wave] = s2; }
  __syncthreads();
  float ts = 0.f, ts2 = 0.f;
  #pragma unroll
  for (int w = 0; w < 8; w++) { ts += red[0][w]; ts2 += red[1][w]; }
  const float mean = ts / (float)D_;
  const float var  = ts2 / (float)D_ - mean * mean;
  const float inv  = rsqrtf(var + 1e-5f);
  const float* sc = ada + (size_t)b * ADA_ + scale_chunk * D_;
  const float* sh = ada + (size_t)b * ADA_ + shift_chunk * D_;
  unsigned short* o = out + (size_t)row * D_;
  #pragma unroll
  for (int j = 0; j < 4; j++) {
    const int d = threadIdx.x + j * 256;
    o[d] = f2bf((vals[j] - mean) * inv * (1.f + sc[d]) + sh[d]);
  }
}

// ---------------------------------------------------------------------------
// Small helpers
// ---------------------------------------------------------------------------
__global__ void k_rowmm(const float* __restrict__ A, int Mm, int Nn, int Kk,
                        const float* __restrict__ W, const float* __restrict__ bias,
                        float* __restrict__ C, int out_silu)
{
  const int idx = blockIdx.x * blockDim.x + threadIdx.x;
  if (idx >= Mm * Nn) return;
  const int m = idx / Nn, n = idx % Nn;
  float acc = bias ? bias[n] : 0.f;
  for (int k = 0; k < Kk; k++) acc += A[(size_t)m * Kk + k] * W[(size_t)k * Nn + n];
  if (out_silu) acc = siluf_(acc);
  C[idx] = acc;
}

__global__ void k_tfeat(const float* __restrict__ t, const float* __restrict__ rff_w,
                        float* __restrict__ tfeat)
{
  const int idx = blockIdx.x * blockDim.x + threadIdx.x;
  if (idx >= B_ * (RFF_ / 2)) return;
  const int b = idx / (RFF_ / 2), j = idx % (RFF_ / 2);
  const float p = t[b] * rff_w[j];
  tfeat[b * RFF_ + j] = __cosf(p);
  tfeat[b * RFF_ + RFF_ / 2 + j] = __sinf(p);
}

__global__ void k_silu_ew(const float* __restrict__ in, float* __restrict__ out, int n)
{
  const int i = blockIdx.x * blockDim.x + threadIdx.x;
  if (i < n) out[i] = siluf_(in[i]);
}

__global__ void k_cvt(const float* __restrict__ in, unsigned short* __restrict__ out, int n)
{
  const int i = blockIdx.x * blockDim.x + threadIdx.x;
  if (i < n) out[i] = f2bf(in[i]);
}

__global__ void k_add_embed(float* __restrict__ x, const float* __restrict__ embed,
                            const int* __restrict__ disc)
{
  const int idx = blockIdx.x * blockDim.x + threadIdx.x;
  if (idx >= B_ * L_ * D_) return;
  const int d = idx % D_;
  const int row = idx / D_;
  x[idx] += embed[(size_t)disc[row] * D_ + d];
}

__global__ void k_res(float* __restrict__ x, const float* __restrict__ y,
                      const float* __restrict__ ada, int chunk)
{
  const int idx = blockIdx.x * blockDim.x + threadIdx.x;
  if (idx >= B_ * L_ * D_) return;
  const int d = idx % D_;
  const int b = idx / (L_ * D_);
  x[idx] += ada[(size_t)b * ADA_ + chunk * D_ + d] * y[idx];
}

__global__ void k_silu_mul(const float* __restrict__ g, const float* __restrict__ u,
                           unsigned short* __restrict__ f, int n)
{
  const int i = blockIdx.x * blockDim.x + threadIdx.x;
  if (i < n) f[i] = f2bf(siluf_(g[i]) * u[i]);
}

__global__ void k_head1(const float* __restrict__ x,
                        const float* __restrict__ w1, const float* __restrict__ b1,
                        const float* __restrict__ w2, const float* __restrict__ b2,
                        float* __restrict__ o1, float* __restrict__ o2)
{
  const int r = blockIdx.x * blockDim.x + threadIdx.x;
  if (r >= B_ * L_) return;
  float a1 = b1[0], a2 = b2[0];
  const float* xr = x + (size_t)r * D_;
  for (int k = 0; k < D_; k++) { const float v = xr[k]; a1 += v * w1[k]; a2 += v * w2[k]; }
  o1[r] = a1; o2[r] = a2;
}

// ---------------------------------------------------------------------------
extern "C" void kernel_launch(void* const* d_in, const int* in_sizes, int n_in,
                              void* d_out, int out_size, void* d_ws, size_t ws_size,
                              hipStream_t stream)
{
  (void)in_sizes; (void)n_in; (void)out_size; (void)ws_size;
  const float* t       = (const float*)d_in[0];
  const float* cont    = (const float*)d_in[1];
  const int*   disc    = (const int*)d_in[2];
  const unsigned char* pm = (const unsigned char*)d_in[3];
  const float* lbias   = (const float*)d_in[4];
  const float* cont_w  = (const float*)d_in[5];
  const float* cont_b  = (const float*)d_in[6];
  const float* embed   = (const float*)d_in[7];
  const float* rff_w   = (const float*)d_in[8];
  const float* tp_w1   = (const float*)d_in[9];
  const float* tp_b1   = (const float*)d_in[10];
  const float* tp_w2   = (const float*)d_in[11];
  const float* tp_b2   = (const float*)d_in[12];
  const float* ch_w    = (const float*)d_in[69];
  const float* ch_b    = (const float*)d_in[70];
  const float* dh_w    = (const float*)d_in[71];
  const float* dh_b    = (const float*)d_in[72];
  const float* sh_w    = (const float*)d_in[73];
  const float* sh_b    = (const float*)d_in[74];
  const float* del_w   = (const float*)d_in[75];
  const float* del_b   = (const float*)d_in[76];
  float* out = (float*)d_out;

  const int ML = B_ * L_;

  // ---- workspace arena (~203 MB) ----
  float* base = (float*)d_ws;
  size_t off = 0;
  float* x      = base + off; off += (size_t)ML * D_;        // residual stream (fp32)
  float* qkvbuf = base + off; off += (size_t)ML * 3 * D_;    // qkv fp32 (also FF gate out)
  float* ubuf   = base + off; off += (size_t)ML * FF_;       // FF up fp32
  float* tmp    = base + off; off += (size_t)ML * D_;        // attn-out / FF-out fp32
  unsigned short* abuf = (unsigned short*)(base + off); off += (size_t)ML * FF_ / 2; // bf16 A operand
  unsigned short* Qb = (unsigned short*)(base + off); off += (size_t)ML * D_ / 2;
  unsigned short* Kb = (unsigned short*)(base + off); off += (size_t)ML * D_ / 2;
  unsigned short* Vt = (unsigned short*)(base + off); off += (size_t)ML * D_ / 2;
  unsigned short* wtb = (unsigned short*)(base + off); off += 12779520 / 2;  // bf16 weights
  float* tfeat  = base + off; off += B_ * RFF_;
  float* hid    = base + off; off += B_ * D_;
  float* tcond  = base + off; off += B_ * D_;
  float* stc    = base + off; off += B_ * D_;
  float* ada    = base + off; off += B_ * ADA_;

  // per-layer transposed bf16 weight slots (padded to 128-row multiples)
  unsigned short* wQkvT  = wtb;                       // 3072 x 1024
  unsigned short* wAoT   = wtb + 3145728;             // 1024 x 1024
  unsigned short* wGateT = wtb + 4194304;             // 2816(pad) x 1024
  unsigned short* wUpT   = wtb + 7077888;             // 2816(pad) x 1024
  unsigned short* wFoT   = wtb + 9961472;             // 1024 x 2752
  unsigned short* wChT   = wtb;                       // 512 x 1024 (reuse)
  unsigned short* wDhT   = wtb + 524288;              // 128(pad) x 1024

  auto gemm = [&](const unsigned short* A, int lda, const unsigned short* Wt,
                  const float* bias, float* C, int ldc, int M, int N, int K) {
    dim3 grid((N + 127) / 128, (M + 127) / 128);
    k_gemm_bf<<<grid, dim3(256), 0, stream>>>(A, lda, Wt, bias, C, ldc, M, N, K);
  };
  auto wcvt = [&](const float* W, int N, int K, unsigned short* Wt) {
    k_wt<<<dim3(K / 32, N / 32), dim3(32, 8), 0, stream>>>(W, N, K, Wt);
  };

  // ---- embedding: x = cont @ cont_w + cont_b + embed[disc] ----
  wcvt(cont_w, D_, CD_, wtb);                               // Wt [1024][512]
  k_cvt<<<(ML * CD_ + 255) / 256, 256, 0, stream>>>(cont, abuf, ML * CD_);
  gemm(abuf, CD_, wtb, cont_b, x, D_, ML, D_, CD_);
  k_add_embed<<<(ML * D_ + 255) / 256, 256, 0, stream>>>(x, embed, disc);

  // ---- time conditioning ----
  k_tfeat<<<(B_ * (RFF_ / 2) + 255) / 256, 256, 0, stream>>>(t, rff_w, tfeat);
  k_rowmm<<<(B_ * D_ + 255) / 256, 256, 0, stream>>>(tfeat, B_, D_, RFF_, tp_w1, tp_b1, hid, 1);
  k_rowmm<<<(B_ * D_ + 255) / 256, 256, 0, stream>>>(hid, B_, D_, D_, tp_w2, tp_b2, tcond, 0);
  k_silu_ew<<<(B_ * D_ + 255) / 256, 256, 0, stream>>>(tcond, stc, B_ * D_);

  for (int li = 0; li < NL_; li++) {
    const float* qkv_w    = (const float*)d_in[13 + 7 * li + 0];
    const float* attn_out = (const float*)d_in[13 + 7 * li + 1];
    const float* ff_gate  = (const float*)d_in[13 + 7 * li + 2];
    const float* ff_up    = (const float*)d_in[13 + 7 * li + 3];
    const float* ff_out   = (const float*)d_in[13 + 7 * li + 4];
    const float* ada_w    = (const float*)d_in[13 + 7 * li + 5];
    const float* ada_b    = (const float*)d_in[13 + 7 * li + 6];

    // convert this layer's weights to bf16 transposed
    wcvt(qkv_w,    3 * D_, D_,  wQkvT);
    wcvt(attn_out, D_,     D_,  wAoT);
    wcvt(ff_gate,  FF_,    D_,  wGateT);
    wcvt(ff_up,    FF_,    D_,  wUpT);
    wcvt(ff_out,   D_,     FF_, wFoT);

    // adaLN params for this layer
    k_rowmm<<<(B_ * ADA_ + 255) / 256, 256, 0, stream>>>(stc, B_, ADA_, D_, ada_w, ada_b, ada, 0);

    // attention branch
    k_ln_mod<<<ML, 256, 0, stream>>>(x, abuf, ada, 0, 1);
    gemm(abuf, D_, wQkvT, nullptr, qkvbuf, 3 * D_, ML, 3 * D_, D_);
    k_rope<<<(B_ * H_ * L_ + 255) / 256, 256, 0, stream>>>(qkvbuf, Qb, Kb, Vt);
    k_attn<<<dim3(L_ / 64, B_ * H_), dim3(128), 0, stream>>>(Qb, Kb, Vt, lbias, pm, abuf);
    gemm(abuf, D_, wAoT, nullptr, tmp, D_, ML, D_, D_);
    k_res<<<(ML * D_ + 255) / 256, 256, 0, stream>>>(x, tmp, ada, 2);

    // FF branch
    k_ln_mod<<<ML, 256, 0, stream>>>(x, abuf, ada, 3, 4);
    gemm(abuf, D_, wGateT, nullptr, qkvbuf, FF_, ML, FF_, D_);
    gemm(abuf, D_, wUpT,   nullptr, ubuf,   FF_, ML, FF_, D_);
    k_silu_mul<<<(ML * FF_ + 255) / 256, 256, 0, stream>>>(qkvbuf, ubuf, abuf, ML * FF_);
    gemm(abuf, FF_, wFoT, nullptr, tmp, D_, ML, D_, FF_);
    k_res<<<(ML * D_ + 255) / 256, 256, 0, stream>>>(x, tmp, ada, 5);
  }

  // ---- output heads ----
  wcvt(ch_w, CD_, D_, wChT);
  wcvt(dh_w, KV_, D_, wDhT);
  k_cvt<<<(ML * D_ + 255) / 256, 256, 0, stream>>>(x, abuf, ML * D_);
  gemm(abuf, D_, wChT, ch_b, out, CD_, ML, CD_, D_);
  gemm(abuf, D_, wDhT, dh_b, out + (size_t)ML * CD_, KV_, ML, KV_, D_);
  k_head1<<<(ML + 255) / 256, 256, 0, stream>>>(
      x, sh_w, sh_b, del_w, del_b,
      out + (size_t)ML * (CD_ + KV_),
      out + (size_t)ML * (CD_ + KV_) + ML);
}
